// TransformerBlock_15058155340446
// MI455X (gfx1250) — matrix operand order repaired and checked by reading, not compiled, not run on hardware
//
#include <hip/hip_runtime.h>
#include <hip/hip_bf16.h>
#include <math.h>

// ---------------- constants ----------------
#define N_EMBD   1024
#define N_HEAD   16
#define HEAD_DIM 64
#define LN_EPS   1e-5f
#define FRAG_LD  24   // bf16 elems per lane row in frag-order storage (16 used + 8 pad = 48B)

typedef __attribute__((ext_vector_type(16))) __bf16       v16bf;
typedef __attribute__((ext_vector_type(8)))  float        v8f;
typedef __attribute__((ext_vector_type(4)))  unsigned int u32x4;
typedef __attribute__((ext_vector_type(8)))  int          i32x8;
typedef __attribute__((ext_vector_type(4)))  int          i32x4;

#define HAVE_TDM (__has_builtin(__builtin_amdgcn_tensor_load_to_lds) && \
                  __has_builtin(__builtin_amdgcn_s_wait_tensorcnt))

// ---------------- WMMA helper ----------------
__device__ __forceinline__ v8f wmma_bf16(v16bf a, v16bf b, v8f c) {
  return __builtin_amdgcn_wmma_f32_16x16x32_bf16(
      false, a, false, b, (short)0, c, false, false);
}

// Load a 16-element bf16 fragment as two 16B vectors (lane-specific base, 16B aligned).
__device__ __forceinline__ v16bf fragload(const __bf16* lanebase) {
  union { v16bf v; u32x4 q[2]; } fr;
  fr.q[0] = *(const u32x4*)(lanebase);
  fr.q[1] = *(const u32x4*)(lanebase + 8);
  return fr.v;
}

__device__ __forceinline__ v16bf fragcol(const __bf16* colbase, int ld, int h) {
  v16bf r;
#pragma unroll
  for (int s = 0; s < 16; ++s) {
    int k = (s < 8) ? (8 * h + s) : (16 + 8 * h + (s - 8));
    r[s] = colbase[k * ld];
  }
  return r;
}

// A-operand fragment-order index for element (m in 0..15, k in 0..31):
// lane = m + 16*((k>>3)&1); slot = (k&7) + 8*((k>>4)&1)   (per ISA 16-bit A 16x32 table)
__device__ __forceinline__ int frag_idx(int m, int k) {
  return (m + ((k & 8) ? 16 : 0)) * FRAG_LD + (k & 7) + ((k & 16) ? 8 : 0);
}

__device__ __forceinline__ void wait_tdm() {
#if HAVE_TDM
  __builtin_amdgcn_s_wait_tensorcnt(0);
#endif
}

// ---------------- weight conversion ----------------
__global__ void k_f32_to_bf16(const float* __restrict__ in, __bf16* __restrict__ out, int n) {
  int i = blockIdx.x * 256 + threadIdx.x;
  if (i < n) out[i] = (__bf16)in[i];
}

// ---------------- layernorm -> bf16 ----------------
__global__ void k_ln(const float* __restrict__ x, const float* __restrict__ g,
                     const float* __restrict__ b, __bf16* __restrict__ out) {
  const int C = N_EMBD;
  const int row = blockIdx.x;
  const float* xr = x + (size_t)row * C;
  __shared__ float red[256];
  float s = 0.f;
  for (int c = threadIdx.x; c < C; c += 256) s += xr[c];
  red[threadIdx.x] = s;
  __syncthreads();
  for (int o = 128; o > 0; o >>= 1) {
    if (threadIdx.x < o) red[threadIdx.x] += red[threadIdx.x + o];
    __syncthreads();
  }
  const float mu = red[0] * (1.f / C);
  __syncthreads();
  float s2 = 0.f;
  for (int c = threadIdx.x; c < C; c += 256) { float d = xr[c] - mu; s2 += d * d; }
  red[threadIdx.x] = s2;
  __syncthreads();
  for (int o = 128; o > 0; o >>= 1) {
    if (threadIdx.x < o) red[threadIdx.x] += red[threadIdx.x + o];
    __syncthreads();
  }
  const float rstd = rsqrtf(red[0] * (1.f / C) + LN_EPS);
  __bf16* orow = out + (size_t)row * C;
  for (int c = threadIdx.x; c < C; c += 256)
    orow[c] = (__bf16)((xr[c] - mu) * rstd * g[c] + b[c]);
}

// ---------------- fused GEMM: out = A[M,K]@B[K,N] + bias (+GELU) (+residual) ----------------
// 128x128x32 block tile, 256 threads = 8 waves, double-buffered k pipeline.
// B tiles staged by the Tensor Data Mover (TDM); A tiles staged in WMMA fragment
// order via vectorized LDS stores.
__global__ void k_gemm(const __bf16* __restrict__ A, const __bf16* __restrict__ Bm,
                       const float* __restrict__ bias, const float* __restrict__ residual,
                       float* __restrict__ outF, __bf16* __restrict__ outB,
                       int M, int N, int K, int gelu) {
  __shared__ __attribute__((aligned(16))) __bf16 Af[2][8][32 * FRAG_LD];
  __shared__ __attribute__((aligned(16))) __bf16 Bs[2][32][136];  // 128 elems + 8 pad per row
  const int tid  = threadIdx.x;
  const int wid  = tid >> 5, lane = tid & 31;
  const int mq   = wid & 3;        // m band-pair
  const int nh   = wid >> 2;       // n tile-quad
  const int bm   = blockIdx.y * 128, bn = blockIdx.x * 128;
  const int steps = K >> 5;

  v8f acc[2][4] = {{{}, {}, {}, {}}, {{}, {}, {}, {}}};

  const int ar = tid >> 1;            // A stage: row 0..127
  const int ak = (tid & 1) * 16;      //          k base 0/16
  const int aband = ar >> 4, am = ar & 15;
#if !HAVE_TDM
  const int br = tid >> 3;            // B stage fallback: row 0..31
  const int bc = (tid & 7) * 16;      //                   col base
#endif

  auto stageA = [&](int i, int p) {
    const __bf16* ag = A + (size_t)(bm + ar) * K + i * 32 + ak;
#pragma unroll
    for (int g = 0; g < 2; ++g) {
      int k0 = ak + g * 8;
      int dlane = am + ((k0 & 8) ? 16 : 0);
      int e0    = (k0 & 16) ? 8 : 0;
      *(u32x4*)&Af[p][aband][dlane * FRAG_LD + e0] = *(const u32x4*)(ag + g * 8);
    }
  };

  auto stageB = [&](int i, int p) {
#if HAVE_TDM
    if (wid == 0) {
      // D# for a 2D bf16 tile: 32 rows x 128 cols, global row stride N elems,
      // LDS pad 4 dwords after every 64 dwords (-> 272B LDS row pitch).
      unsigned lds = (unsigned)(size_t)(void*)&Bs[p][0][0];   // addr[31:0] = LDS offset
      unsigned long long ga =
          (unsigned long long)(size_t)(Bm + (size_t)i * 32 * N + bn);
      u32x4 g0 = { 1u,                                  // count=1 (valid user descriptor)
                   lds,                                 // lds_addr
                   (unsigned)(ga & 0xFFFFFFFFu),        // global_addr[31:0]
                   (unsigned)((ga >> 32) & 0x01FFFFFFu) // global_addr[56:32]
                     | (2u << 30) };                    // type=2 ("image")
      i32x8 g1 = { (int)((1u << 16)                     // data_size = 2 bytes
                     | (1u << 20)                       // pad_enable
                     | (5u << 22)                       // pad_interval: 64 dwords
                     | (3u << 25)),                     // pad_amount: 4 dwords
                   (int)(128u << 16),                   // tensor_dim0 = 128
                   (int)(32u << 16),                    // tensor_dim1 = 32
                   (int)(128u << 16),                   // tile_dim0 = 128
                   32,                                  // tile_dim1 = 32 (tile_dim2 = 0)
                   N,                                   // tensor_dim0_stride = N elems
                   0, 0 };
      i32x4 g2 = {0, 0, 0, 0};
      i32x4 g3 = {0, 0, 0, 0};
      i32x8 g4 = {0, 0, 0, 0, 0, 0, 0, 0};
      // clang-23 / therock-10.0 lane: 6-arg form (extra int32x8 before cpol)
      __builtin_amdgcn_tensor_load_to_lds(g0, g1, g2, g3, g4, 0);
    }
#else
    const __bf16* bg = Bm + (size_t)(i * 32 + br) * N + bn + bc;
    *(u32x4*)&Bs[p][br][bc]     = *(const u32x4*)(bg);
    *(u32x4*)&Bs[p][br][bc + 8] = *(const u32x4*)(bg + 8);
#endif
  };

  // prologue: fill buffer 0
  stageA(0, 0);
  stageB(0, 0);

  for (int i = 0; i < steps; ++i) {
    const int p = i & 1;
    wait_tdm();            // wave0: TENSORcnt==0 -> Bs[p] resident (others: no-op)
    __syncthreads();       // publish Bs[p]/Af[p]; retire last iter's reads of buffer 1-p
    if (i + 1 < steps) {   // prefetch next tile into the other buffer, overlapped w/ WMMA
      stageA(i + 1, 1 - p);
      stageB(i + 1, 1 - p);
    }
    v16bf a0 = fragload(&Af[p][2 * mq + 0][lane * FRAG_LD]);
    v16bf a1 = fragload(&Af[p][2 * mq + 1][lane * FRAG_LD]);
#pragma unroll
    for (int nj = 0; nj < 4; ++nj) {
      v16bf bf = fragcol(&Bs[p][0][(nh * 4 + nj) * 16 + (lane & 15)], 136, lane >> 4);  // lane = N col
      acc[0][nj] = wmma_bf16(a0, bf, acc[0][nj]);
      acc[1][nj] = wmma_bf16(a1, bf, acc[1][nj]);
    }
  }

  const int nl = lane & 15, half = lane >> 4;
#pragma unroll
  for (int mi = 0; mi < 2; ++mi)
#pragma unroll
    for (int nj = 0; nj < 4; ++nj)
#pragma unroll
      for (int r = 0; r < 8; ++r) {
        const int gm = bm + (2 * mq + mi) * 16 + r + 8 * half;
        const int gn = bn + (nh * 4 + nj) * 16 + nl;
        float v = acc[mi][nj][r] + (bias ? bias[gn] : 0.f);
        if (gelu) v = 0.5f * v * (1.f + erff(v * 0.70710678118f));
        if (residual) v += residual[(size_t)gm * N + gn];
        if (outF) outF[(size_t)gm * N + gn] = v;
        else      outB[(size_t)gm * N + gn] = (__bf16)v;
      }
}

// ---------------- flash attention (causal): block = 64 q rows of one (b,h) ----------------
// 128 threads = 4 waves; wave wq owns q rows [16*wq, +16).
__global__ void k_attn(const __bf16* __restrict__ qkv, __bf16* __restrict__ o) {
  const int T = 2048, C = N_EMBD;
  const int qb = blockIdx.x, h = blockIdx.y, b = blockIdx.z;
  const int q0 = qb * 64;
  const int tid = threadIdx.x, wq = tid >> 5, lane = tid & 31;
  const int nl = lane & 15, half = lane >> 4;

  __shared__ __attribute__((aligned(16))) __bf16 Qf[4][2][32 * FRAG_LD];
  __shared__ __attribute__((aligned(16))) __bf16 Kt[64][72]; // K^T: row=d, col=k
  __shared__ __attribute__((aligned(16))) __bf16 Vs[64][72]; // V:   row=k, col=d
  __shared__ __attribute__((aligned(16))) __bf16 Pf[4][2][32 * FRAG_LD];

  const size_t rs = 3 * C;
  const __bf16* qbase = qkv + ((size_t)b * T) * rs + h * HEAD_DIM;
  const __bf16* kbase = qbase + C;
  const __bf16* vbase = qbase + 2 * C;

  { // stage Q (64x64) into fragment order, vectorized
    int r = tid >> 1, c0 = (tid & 1) * 32;
    int band = r >> 4, m = r & 15;
    const __bf16* src = qbase + (size_t)(q0 + r) * rs + c0;
#pragma unroll
    for (int g = 0; g < 4; ++g) {
      int d = c0 + g * 8, c = d >> 5, d2 = d & 31;
      int dlane = m + ((d2 & 8) ? 16 : 0);
      int e0    = (d2 & 16) ? 8 : 0;
      *(u32x4*)&Qf[band][c][dlane * FRAG_LD + e0] = *(const u32x4*)(src + g * 8);
    }
  }
  __syncthreads();
  v16bf qa0 = fragload(&Qf[wq][0][lane * FRAG_LD]);
  v16bf qa1 = fragload(&Qf[wq][1][lane * FRAG_LD]);

  float mrow[8], lrow[8];
  v8f oacc[4] = {{}, {}, {}, {}};
#pragma unroll
  for (int r = 0; r < 8; ++r) { mrow[r] = -INFINITY; lrow[r] = 0.f; }

  const float scale = 0.125f; // 1/sqrt(64)
  for (int kt = 0; kt <= qb; ++kt) {
    { // stage K transposed (b16 scatter at store time) + V row-major (vectorized)
      int r = tid >> 1, c0 = (tid & 1) * 32;
      const __bf16* ks = kbase + (size_t)(kt * 64 + r) * rs + c0;
      const __bf16* vs = vbase + (size_t)(kt * 64 + r) * rs + c0;
      union { u32x4 q[4]; __bf16 hh[32]; } kv;
#pragma unroll
      for (int g = 0; g < 4; ++g) kv.q[g] = *(const u32x4*)(ks + g * 8);
#pragma unroll
      for (int i = 0; i < 32; ++i) Kt[c0 + i][r] = kv.hh[i];
#pragma unroll
      for (int g = 0; g < 4; ++g) *(u32x4*)&Vs[r][c0 + g * 8] = *(const u32x4*)(vs + g * 8);
    }
    __syncthreads();

    // S = Q K^T : B fragments read down Kt columns (lane = key col)
    v8f S[4];
#pragma unroll
    for (int nt = 0; nt < 4; ++nt) {
      v8f z = {};
      z = wmma_bf16(qa0, fragcol(&Kt[0][nt * 16 + nl], 72, half), z);
      z = wmma_bf16(qa1, fragcol(&Kt[32][nt * 16 + nl], 72, half), z);
      S[nt] = z;
    }

    const bool diag = (kt == qb);
#pragma unroll
    for (int nt = 0; nt < 4; ++nt)
#pragma unroll
      for (int r = 0; r < 8; ++r) {
        float v = S[nt][r] * scale;
        if (diag) {
          int qg = q0 + wq * 16 + r + 8 * half;
          int kg = kt * 64 + nt * 16 + nl;
          if (kg > qg) v = -INFINITY;
        }
        S[nt][r] = v;
      }

    // online softmax (row stats replicated across each 16-lane half)
    float alpha[8];
#pragma unroll
    for (int r = 0; r < 8; ++r) {
      float rm = fmaxf(fmaxf(S[0][r], S[1][r]), fmaxf(S[2][r], S[3][r]));
#pragma unroll
      for (int off = 1; off < 16; off <<= 1) rm = fmaxf(rm, __shfl_xor(rm, off, 32));
      float mnew = fmaxf(mrow[r], rm);
      alpha[r] = __expf(mrow[r] - mnew);
      float rsum = 0.f;
#pragma unroll
      for (int nt = 0; nt < 4; ++nt) {
        float p = __expf(S[nt][r] - mnew);
        S[nt][r] = p;
        rsum += p;
      }
#pragma unroll
      for (int off = 1; off < 16; off <<= 1) rsum += __shfl_xor(rsum, off, 32);
      lrow[r] = lrow[r] * alpha[r] + rsum;
      mrow[r] = mnew;
    }
#pragma unroll
    for (int dt = 0; dt < 4; ++dt)
#pragma unroll
      for (int r = 0; r < 8; ++r) oacc[dt][r] *= alpha[r];

    // P: C-layout -> fragment order (per-wave private region)
#pragma unroll
    for (int nt = 0; nt < 4; ++nt)
#pragma unroll
      for (int r = 0; r < 8; ++r) {
        int k = nt * 16 + nl, c = k >> 5, k2 = k & 31;
        Pf[wq][c][frag_idx(r + 8 * half, k2)] = (__bf16)S[nt][r];
      }
    __syncthreads();

    v16bf pa0 = fragload(&Pf[wq][0][lane * FRAG_LD]);
    v16bf pa1 = fragload(&Pf[wq][1][lane * FRAG_LD]);
#pragma unroll
    for (int dt = 0; dt < 4; ++dt) {
      oacc[dt] = wmma_bf16(pa0, fragcol(&Vs[0][dt * 16 + nl], 72, half), oacc[dt]);
      oacc[dt] = wmma_bf16(pa1, fragcol(&Vs[32][dt * 16 + nl], 72, half), oacc[dt]);
    }
    __syncthreads();
  }

  // normalize, write O
#pragma unroll
  for (int dt = 0; dt < 4; ++dt)
#pragma unroll
    for (int r = 0; r < 8; ++r) {
      int q = q0 + wq * 16 + r + 8 * half;
      float val = oacc[dt][r] / lrow[r];
      o[((size_t)b * T + q) * C + h * HEAD_DIM + dt * 16 + nl] = (__bf16)val;
    }
}

// ---------------- launch ----------------
extern "C" void kernel_launch(void* const* d_in, const int* in_sizes, int n_in,
                              void* d_out, int out_size, void* d_ws, size_t ws_size,
                              hipStream_t stream) {
  (void)in_sizes; (void)n_in; (void)out_size; (void)ws_size;
  const int Bb = 2, T = 2048, C = N_EMBD;
  const int M = Bb * T; // 4096

  const float* x     = (const float*)d_in[0];
  const float* ln1_g = (const float*)d_in[1];
  const float* ln1_b = (const float*)d_in[2];
  const float* qkv_w = (const float*)d_in[3];
  const float* qkv_b = (const float*)d_in[4];
  const float* out_w = (const float*)d_in[5];
  const float* out_b = (const float*)d_in[6];
  const float* ln2_g = (const float*)d_in[7];
  const float* ln2_b = (const float*)d_in[8];
  const float* fc1_w = (const float*)d_in[9];
  const float* fc1_b = (const float*)d_in[10];
  const float* fc2_w = (const float*)d_in[11];
  const float* fc2_b = (const float*)d_in[12];

  char* ws = (char*)d_ws;
  size_t off = 0;
  auto take = [&](size_t elems, size_t esz) -> void* {
    void* p = ws + off;
    off += (elems * esz + 255) & ~(size_t)255;
    return p;
  };
  __bf16* wqkv  = (__bf16*)take((size_t)C * 3 * C, 2);
  __bf16* wout  = (__bf16*)take((size_t)C * C, 2);
  __bf16* wfc1  = (__bf16*)take((size_t)C * 4 * C, 2);
  __bf16* wfc2  = (__bf16*)take((size_t)4 * C * C, 2);
  __bf16* hbf   = (__bf16*)take((size_t)M * C, 2);
  __bf16* qkvbf = (__bf16*)take((size_t)M * 3 * C, 2);
  __bf16* obf   = (__bf16*)take((size_t)M * C, 2);
  __bf16* hff   = (__bf16*)take((size_t)M * 4 * C, 2);
  float*  x1    = (float*) take((size_t)M * C, 4);

  // 1) weights -> bf16 (fit entirely in 192MB L2; reused by all 32 M-blocks)
  k_f32_to_bf16<<<(3 * C * C) / 256, 256, 0, stream>>>(qkv_w, wqkv, 3 * C * C);
  k_f32_to_bf16<<<(C * C) / 256,     256, 0, stream>>>(out_w, wout, C * C);
  k_f32_to_bf16<<<(4 * C * C) / 256, 256, 0, stream>>>(fc1_w, wfc1, 4 * C * C);
  k_f32_to_bf16<<<(4 * C * C) / 256, 256, 0, stream>>>(fc2_w, wfc2, 4 * C * C);

  // 2) LN1
  k_ln<<<M, 256, 0, stream>>>(x, ln1_g, ln1_b, hbf);
  // 3) QKV
  k_gemm<<<dim3(3 * C / 128, M / 128), 256, 0, stream>>>(hbf, wqkv, qkv_b, nullptr,
                                                         nullptr, qkvbf, M, 3 * C, C, 0);
  // 4) causal flash attention
  k_attn<<<dim3(T / 64, N_HEAD, Bb), 128, 0, stream>>>(qkvbf, obf);
  // 5) x1 = x + o @ out_w + out_b
  k_gemm<<<dim3(C / 128, M / 128), 256, 0, stream>>>(obf, wout, out_b, x,
                                                     x1, nullptr, M, C, C, 0);
  // 6) LN2
  k_ln<<<M, 256, 0, stream>>>(x1, ln2_g, ln2_b, hbf);
  // 7) hff = gelu(h @ fc1_w + fc1_b)
  k_gemm<<<dim3(4 * C / 128, M / 128), 256, 0, stream>>>(hbf, wfc1, fc1_b, nullptr,
                                                         nullptr, hff, M, 4 * C, C, 1);
  // 8) out = x1 + hff @ fc2_w + fc2_b
  k_gemm<<<dim3(C / 128, M / 128), 256, 0, stream>>>(hff, wfc2, fc2_b, x1,
                                                     (float*)d_out, nullptr, M, C, 4 * C, 0);
}
